// GraphModel_41111426957574
// MI455X (gfx1250) — matrix operand order repaired and checked by reading, not compiled, not run on hardware
//
#include <hip/hip_runtime.h>
#include <math.h>

#define NN 50000      // nodes
#define NE 800000     // edges
#define NG 256        // graphs
#define DIN 256
#define DH 512

typedef __bf16 bf16_t;
typedef __attribute__((ext_vector_type(16))) __bf16 v16bf;
typedef __attribute__((ext_vector_type(8)))  __bf16 v8bf;
typedef __attribute__((ext_vector_type(8)))  float  v8f;

// ---------------- init: deg=1 (self loop), zero graph sums/counts ----------
__global__ void k_init(float* __restrict__ dinv, float* __restrict__ gsums,
                       int* __restrict__ counts) {
  int i = blockIdx.x * blockDim.x + threadIdx.x;
  if (i < NN) dinv[i] = 1.0f;
  if (i < NG * DH) gsums[i] = 0.0f;
  if (i < NG) counts[i] = 0;
}

// ---------------- degree count over edge destinations ----------------------
__global__ void k_deg(const int* __restrict__ dst, float* __restrict__ deg) {
  int e = blockIdx.x * blockDim.x + threadIdx.x;
  if (e < NE) atomicAdd(&deg[dst[e]], 1.0f);
}

__global__ void k_deg_fin(float* __restrict__ dinv) {
  int i = blockIdx.x * blockDim.x + threadIdx.x;
  if (i < NN) dinv[i] = rsqrtf(dinv[i]);   // deg >= 1 thanks to self loop
}

// ---------------- f32 -> bf16 convert (optionally fused ReLU) --------------
__global__ void k_cvt(const float* __restrict__ s, bf16_t* __restrict__ d,
                      long long n, int relu) {
  long long i = (long long)blockIdx.x * blockDim.x + threadIdx.x;
  if (i < n) {
    float v = s[i];
    if (relu) v = fmaxf(v, 0.0f);
    d[i] = (bf16_t)v;
  }
}

// ---------------- W[K][N] f32 -> Wt[N][K] bf16 -----------------------------
__global__ void k_wt(const float* __restrict__ W, bf16_t* __restrict__ Wt,
                     int K, int N) {
  int idx = blockIdx.x * blockDim.x + threadIdx.x;
  if (idx < K * N) {
    int n = idx / K, k = idx - n * K;
    Wt[idx] = (bf16_t)W[(size_t)k * N + n];
  }
}

// ---------------- bf16 WMMA GEMM: C[M x N] = A[M x K] * Wt[N x K]^T --------
// grid.x = M/16, block = 256 (8 waves), wave w -> columns [64w, 64w+64)
// Requires: N == 512, K % 32 == 0, M % 16 == 0.
__global__ __launch_bounds__(256) void k_gemm_wmma(
    const bf16_t* __restrict__ A, const bf16_t* __restrict__ Wt,
    float* __restrict__ C, int K, int N) {
  const int lane = threadIdx.x & 31;
  const int wave = threadIdx.x >> 5;
  const int m0 = blockIdx.x << 4;
  const int n0 = wave << 6;
  const int l  = lane & 15;
  const int hi = lane >> 4;

  v8f acc[4] = {};
  // A fragment source: lane<16 holds K[0..7]+K[16..23] of row m0+l,
  // lane>=16 holds K[8..15]+K[24..31] of the same row (ISA 16-bit A layout).
  const bf16_t* arow = A + (size_t)(m0 + l) * K + hi * 8;

  for (int k0 = 0; k0 < K; k0 += 32) {
    v8bf alo = *(const v8bf*)(arow + k0);
    v8bf ahi = *(const v8bf*)(arow + k0 + 16);
    v16bf a = __builtin_shufflevector(alo, ahi,
        0, 1, 2, 3, 4, 5, 6, 7, 8, 9, 10, 11, 12, 13, 14, 15);
#pragma unroll
    for (int j = 0; j < 4; ++j) {
      // B fragment: lane&15 selects column, lane<16 holds K[0..7]+K[16..23],
      // lane>=16 holds K[8..15]+K[24..31] (same order as A) -> two 16B loads.
      const bf16_t* brow = Wt + (size_t)(n0 + j * 16 + l) * K + k0 + hi * 8;
      v8bf blo = *(const v8bf*)(brow);
      v8bf bhi = *(const v8bf*)(brow + 16);
      v16bf b = __builtin_shufflevector(blo, bhi,
          0, 1, 2, 3, 4, 5, 6, 7, 8, 9, 10, 11, 12, 13, 14, 15);
      acc[j] = __builtin_amdgcn_wmma_f32_16x16x32_bf16(
          false, a, false, b, (short)0, acc[j], false, false);
    }
  }

  // C/D layout: VGPR r of lane L -> row m0 + r + (L>=16 ? 8 : 0), col n0+(L&15)+16j
  float* cbase = C + (size_t)m0 * N + n0 + l;
#pragma unroll
  for (int r = 0; r < 8; ++r) {
    float* crow = cbase + (size_t)(r + hi * 8) * N;
    crow[0]  = acc[0][r];
    crow[16] = acc[1][r];
    crow[32] = acc[2][r];
    crow[48] = acc[3][r];
  }
}

// ---------------- self-loop term + bias init of aggregation buffer ---------
__global__ void k_agg_self(const float* __restrict__ xw,
                           const float* __restrict__ dinv,
                           const float* __restrict__ bias,
                           float* __restrict__ out) {
  size_t idx = (size_t)blockIdx.x * blockDim.x + threadIdx.x; // NN*DH exact
  int node = (int)(idx >> 9);
  int c = (int)(idx & 511);
  float di = dinv[node];
  out[idx] = xw[idx] * di * di + bias[c];
}

// ---------------- edge scatter-add: out[d] += xw[s] * dinv[s]*dinv[d] ------
// grid.x = NE, block = 256 threads, 2 channels per thread (float2 gather)
__global__ __launch_bounds__(256) void k_agg_edges(
    const float* __restrict__ xw, const int* __restrict__ src,
    const int* __restrict__ dst, const float* __restrict__ dinv,
    float* __restrict__ out) {
  int e = blockIdx.x;
  int s = src[e], d = dst[e];
  float norm = dinv[s] * dinv[d];
  int c = threadIdx.x * 2;
  float2 v = *(const float2*)(xw + ((size_t)s << 9) + c);
  float* o = out + ((size_t)d << 9) + c;
  atomicAdd(o,     v.x * norm);
  atomicAdd(o + 1, v.y * norm);
}

// ---------------- readout ---------------------------------------------------
__global__ void k_count(const int* __restrict__ gb, int* __restrict__ counts) {
  int i = blockIdx.x * blockDim.x + threadIdx.x;
  if (i < NN) atomicAdd(&counts[gb[i]], 1);
}

__global__ void k_gsum(const float* __restrict__ h, const int* __restrict__ gb,
                       float* __restrict__ gsums) {
  size_t idx = (size_t)blockIdx.x * blockDim.x + threadIdx.x; // NN*DH exact
  int node = (int)(idx >> 9);
  int c = (int)(idx & 511);
  atomicAdd(gsums + ((size_t)gb[node] << 9) + c, h[idx]);
}

__global__ void k_maxlen(const int* __restrict__ counts,
                         float* __restrict__ maxlen) {
  __shared__ int sm[256];
  sm[threadIdx.x] = counts[threadIdx.x];   // NG == 256 == blockDim
  __syncthreads();
  for (int s = 128; s > 0; s >>= 1) {
    if (threadIdx.x < s) sm[threadIdx.x] = max(sm[threadIdx.x], sm[threadIdx.x + s]);
    __syncthreads();
  }
  if (threadIdx.x == 0) maxlen[0] = (float)sm[0];
}

__global__ void k_final(const float* __restrict__ gsums,
                        const float* __restrict__ maxlen,
                        float* __restrict__ out) {
  int i = blockIdx.x * blockDim.x + threadIdx.x;  // NG*DH exact
  out[i] = tanhf(gsums[i] / maxlen[0]);
}

extern "C" void kernel_launch(void* const* d_in, const int* in_sizes, int n_in,
                              void* d_out, int out_size, void* d_ws, size_t ws_size,
                              hipStream_t stream) {
  const float* x  = (const float*)d_in[0];
  const float* W1 = (const float*)d_in[1];
  const float* b1 = (const float*)d_in[2];
  const float* W2 = (const float*)d_in[3];
  const float* b2 = (const float*)d_in[4];
  const int*   ei = (const int*)d_in[5];
  const int*   gb = (const int*)d_in[6];
  const int* src = ei;
  const int* dst = ei + NE;

  // Workspace layout (all 32B-aligned regions first)
  char* ws = (char*)d_ws;
  float*  bufA   = (float*)ws;                                 // NN*DH f32
  float*  bufB   = bufA + (size_t)NN * DH;                     // NN*DH f32
  bf16_t* bfX    = (bf16_t*)(bufB + (size_t)NN * DH);          // NN*DH bf16 (x, then relu(h))
  bf16_t* bfW    = bfX + (size_t)NN * DH;                      // DH*DH bf16 (Wt)
  float*  dinv   = (float*)(bfW + (size_t)DH * DH);            // NN f32
  float*  gsums  = dinv + NN;                                  // NG*DH f32
  int*    counts = (int*)(gsums + (size_t)NG * DH);            // NG i32
  float*  maxlen = (float*)(counts + NG);                      // 1 f32

  const int B = 256;
  // ---- degrees / dinv -----------------------------------------------------
  k_init   <<<(NG * DH + B - 1) / B, B, 0, stream>>>(dinv, gsums, counts);
  k_deg    <<<(NE + B - 1) / B, B, 0, stream>>>(dst, dinv);
  k_deg_fin<<<(NN + B - 1) / B, B, 0, stream>>>(dinv);

  // ---- layer 1: xw1 = x @ W1 (bf16 WMMA, f32 acc) -------------------------
  k_cvt<<<(NN * DIN) / B, B, 0, stream>>>(x, bfX, (long long)NN * DIN, 0);
  k_wt <<<(DIN * DH + B - 1) / B, B, 0, stream>>>(W1, bfW, DIN, DH);
  k_gemm_wmma<<<NN / 16, B, 0, stream>>>(bfX, bfW, bufA, DIN, DH);
  k_agg_self <<<(NN * DH) / B, B, 0, stream>>>(bufA, dinv, b1, bufB);
  k_agg_edges<<<NE, B, 0, stream>>>(bufA, src, dst, dinv, bufB);

  // ---- layer 2: hw2 = relu(h) @ W2 ---------------------------------------
  k_cvt<<<(NN * DH) / B, B, 0, stream>>>(bufB, bfX, (long long)NN * DH, 1);
  k_wt <<<(DH * DH + B - 1) / B, B, 0, stream>>>(W2, bfW, DH, DH);
  k_gemm_wmma<<<NN / 16, B, 0, stream>>>(bfX, bfW, bufA, DH, DH);
  k_agg_self <<<(NN * DH) / B, B, 0, stream>>>(bufA, dinv, b2, bufB);
  k_agg_edges<<<NE, B, 0, stream>>>(bufA, src, dst, dinv, bufB);

  // ---- readout ------------------------------------------------------------
  k_count <<<(NN + B - 1) / B, B, 0, stream>>>(gb, counts);
  k_gsum  <<<(NN * DH) / B, B, 0, stream>>>(bufB, gb, gsums);
  k_maxlen<<<1, 256, 0, stream>>>(counts, maxlen);
  k_final <<<(NG * DH) / B, B, 0, stream>>>(gsums, maxlen, (float*)d_out);
}
